// FADocker_19404662243785
// MI455X (gfx1250) — matrix operand order repaired and checked by reading, not compiled, not run on hardware
//
#include <hip/hip_runtime.h>
#include <hip/hip_bf16.h>

#define B_ 2
#define L_ 384
#define H_ 256
#define A_ 14
#define NT 24          // L_/16 j-tiles
#define JS 4           // j-range split per (b,i) for occupancy
#define TPJ (NT / JS)  // tiles per wave = 6 (processed 2 at a time)
#define WPB 4          // waves per block

typedef __attribute__((ext_vector_type(16))) _Float16 v16h;
typedef __attribute__((ext_vector_type(8)))  float    v8f;

// ---------------------------------------------------------------------------
// Kernel 1: hw = h @ Wx^T + b ; hu = h @ Ux^T + b   (fp32, cheap: 0.2 GFLOP)
// ---------------------------------------------------------------------------
__global__ void __launch_bounds__(256)
proj_kernel(const float* __restrict__ h,
            const float* __restrict__ Ww, const float* __restrict__ Wb,
            const float* __restrict__ Uw, const float* __restrict__ Ub,
            float* __restrict__ hw, float* __restrict__ hu)
{
    __shared__ __align__(16) float sh[H_];
    const int row = blockIdx.x;        // 0 .. B_*L_-1
    const int t   = threadIdx.x;       // output column 0..255
    sh[t] = h[(size_t)row * H_ + t];
    __syncthreads();

    const float4* h4 = (const float4*)sh;
    const float4* w4 = (const float4*)(Ww + (size_t)t * H_);
    const float4* u4 = (const float4*)(Uw + (size_t)t * H_);
    float aw = Wb[t];
    float au = Ub[t];
#pragma unroll 8
    for (int k = 0; k < H_ / 4; ++k) {
        const float4 hv = h4[k];
        const float4 wv = w4[k];
        const float4 uv = u4[k];
        aw += hv.x * wv.x + hv.y * wv.y + hv.z * wv.z + hv.w * wv.w;
        au += hv.x * uv.x + hv.y * uv.y + hv.z * uv.z + hv.w * uv.w;
    }
    hw[(size_t)row * H_ + t] = aw;
    hu[(size_t)row * H_ + t] = au;
}

// ---------------------------------------------------------------------------
// Kernel 2 (WMMA): per wave, fixed (b, i, j-slice). Two j-tiles in flight:
// two independent v_wmma_f32_16x16x32_f16 accumulator chains so tile-B's
// A-build VALU work fills tile-A's WMMA hazard window.
// ---------------------------------------------------------------------------
template <int BASE>
__device__ inline void pack_relu4(v16h& av, float4 u, float4 w)
{
    av[BASE + 0] = (_Float16)fmaxf(u.x + w.x, 0.f);
    av[BASE + 1] = (_Float16)fmaxf(u.y + w.y, 0.f);
    av[BASE + 2] = (_Float16)fmaxf(u.z + w.z, 0.f);
    av[BASE + 3] = (_Float16)fmaxf(u.w + w.w, 0.f);
}

__device__ inline v16h build_a_tile(const float* __restrict__ hub, int kb,
                                    const float* __restrict__ shw)
{
    const float4 u0 = *(const float4*)(hub + kb);
    const float4 u1 = *(const float4*)(hub + kb + 4);
    const float4 u2 = *(const float4*)(hub + kb + 16);
    const float4 u3 = *(const float4*)(hub + kb + 20);
    const float4 w0 = *(const float4*)(shw + kb);
    const float4 w1 = *(const float4*)(shw + kb + 4);
    const float4 w2 = *(const float4*)(shw + kb + 16);
    const float4 w3 = *(const float4*)(shw + kb + 20);
    v16h av;
    pack_relu4<0>(av, u0, w0);
    pack_relu4<4>(av, u1, w1);
    pack_relu4<8>(av, u2, w2);
    pack_relu4<12>(av, u3, w3);
    return av;
}

__global__ void __launch_bounds__(WPB * 32)
gate_acc_kernel(const float* __restrict__ hw, const float* __restrict__ hu,
                const float* __restrict__ Tw, const float* __restrict__ Tb,
                const float* __restrict__ X,  const int* __restrict__ mask,
                float* __restrict__ f)
{
    __shared__ __align__(16) float s_hw[WPB][H_];

    const int lane = threadIdx.x & 31;
    const int wv   = threadIdx.x >> 5;
    const int wid  = blockIdx.x * WPB + wv;        // 0 .. B_*L_*JS-1
    const int js   = wid % JS;
    const int ig   = wid / JS;                     // 0 .. B_*L_-1
    const int b    = ig / L_;
    const int col  = lane & 15;  // a-column (B/C/D) == j-row within tile (A)
    const int hi   = lane >> 4;  // lane half
    const int ahk  = hi * 8;     // A-matrix K offset within a 32-chunk
    const int bhk  = hi * 8;     // B-matrix K offset within a 32-chunk

    // Stage this wave's hw row (1 KB) into LDS: chunk reads become broadcasts
    // (compiler then keeps them in VGPRs across the j loop).
    const float* hwrow = hw + (size_t)ig * H_;
    for (int t = lane; t < H_; t += 32) s_hw[wv][t] = hwrow[t];
    __syncthreads();

    // Preload B = Tx_w^T as f16 (8 K-chunks of 32). K = 8*half + (e&7) + 16*(e>>3) per lane
    // half, column = lane&15; a >= 14 zero-padded.
    v16h Breg[8];
#pragma unroll
    for (int ck = 0; ck < 8; ++ck) {
        v16h v;
#pragma unroll
        for (int e = 0; e < 16; ++e) {
            const int k  = ck * 32 + bhk + (e & 7) + (e & 8) * 2;
            const float tv = (col < A_) ? Tw[col * H_ + k] : 0.f;
            v[e] = (_Float16)tv;
        }
        Breg[ck] = v;
    }

    const float tb = (col < A_) ? Tb[col] : 0.f;
    float xi0 = 0.f, xi1 = 0.f, xi2 = 0.f;
    if (col < A_) {
        const float* xp = X + ((size_t)ig * A_ + col) * 3;
        xi0 = xp[0]; xi1 = xp[1]; xi2 = xp[2];
    }

    float acc0 = 0.f, acc1 = 0.f, acc2 = 0.f;
    const int jt0 = js * TPJ;
    for (int jt = jt0; jt < jt0 + TPJ; jt += 2) {
        // Two tiles in flight: rows jt*16+col and (jt+1)*16+col.
        const float* hubA = hu + ((size_t)b * L_ + jt * 16 + col) * H_;
        const float* hubB = hubA + 16 * H_;
        v8f c0 = {};
        v8f c1 = {};
#pragma unroll
        for (int ck = 0; ck < 8; ++ck) {
            const int kb = ck * 32 + ahk;
            const v16h av0 = build_a_tile(hubA, kb, &s_hw[wv][0]);
            const v16h av1 = build_a_tile(hubB, kb, &s_hw[wv][0]);
            c0 = __builtin_amdgcn_wmma_f32_16x16x32_f16(
                     false, av0, false, Breg[ck], (short)0, c0, false, false);
            c1 = __builtin_amdgcn_wmma_f32_16x16x32_f16(
                     false, av1, false, Breg[ck], (short)0, c1, false, false);
        }
        // C/D layout: VGPR r -> M = r (lanes 0-15) / r+8 (lanes 16-31), N = col.
        if (col < A_) {
#pragma unroll
            for (int m = 0; m < 8; ++m) {
                const int   j0 = jt * 16 + m + hi * 8;
                const int   j1 = j0 + 16;
                const float g0 = (c0[m] + tb) * (float)mask[b * L_ + j0];
                const float g1 = (c1[m] + tb) * (float)mask[b * L_ + j1];
                const float* xj0 = X + (((size_t)b * L_ + j0) * A_ + col) * 3;
                const float* xj1 = X + (((size_t)b * L_ + j1) * A_ + col) * 3;
                acc0 += g0 * (xi0 - xj0[0]) + g1 * (xi0 - xj1[0]);
                acc1 += g0 * (xi1 - xj0[1]) + g1 * (xi1 - xj1[1]);
                acc2 += g0 * (xi2 - xj0[2]) + g1 * (xi2 - xj1[2]);
            }
        }
    }

    if (col < A_) {
        float* fp = f + ((size_t)ig * A_ + col) * 3;
        atomicAdd(fp + 0, acc0);
        atomicAdd(fp + 1, acc1);
        atomicAdd(fp + 2, acc2);
    }
}

// ---------------------------------------------------------------------------
// Kernel 3: out = X + clip(f / (1e-6 + sum_j mask), -20, 20)
// ---------------------------------------------------------------------------
__global__ void __launch_bounds__(256)
finalize_kernel(const float* __restrict__ X, const int* __restrict__ mask,
                const float* __restrict__ f, float* __restrict__ out, int n)
{
    const int idx = blockIdx.x * blockDim.x + threadIdx.x;
    if (idx >= n) return;
    const int b = idx / (L_ * A_ * 3);
    float s = 0.f;
    for (int j = 0; j < L_; ++j) s += (float)mask[b * L_ + j];
    float v = f[idx] / (1e-6f + s);
    v = fminf(fmaxf(v, -20.f), 20.f);
    out[idx] = X[idx] + v;
}

// ---------------------------------------------------------------------------
extern "C" void kernel_launch(void* const* d_in, const int* in_sizes, int n_in,
                              void* d_out, int out_size, void* d_ws, size_t ws_size,
                              hipStream_t stream)
{
    const float* h    = (const float*)d_in[0];
    const float* X    = (const float*)d_in[1];
    const int*   mask = (const int*)  d_in[2];
    const float* Wxw  = (const float*)d_in[3];
    const float* Wxb  = (const float*)d_in[4];
    const float* Uxw  = (const float*)d_in[5];
    const float* Uxb  = (const float*)d_in[6];
    const float* Txw  = (const float*)d_in[7];
    const float* Txb  = (const float*)d_in[8];
    float* out = (float*)d_out;

    float* hw = (float*)d_ws;                    // B*L*H
    float* hu = hw + (size_t)B_ * L_ * H_;       // B*L*H
    float* f  = hu + (size_t)B_ * L_ * H_;       // B*L*A*3

    const int n_out = B_ * L_ * A_ * 3;
    hipMemsetAsync(f, 0, sizeof(float) * (size_t)n_out, stream);

    proj_kernel<<<B_ * L_, 256, 0, stream>>>(h, Wxw, Wxb, Uxw, Uxb, hw, hu);

    gate_acc_kernel<<<(B_ * L_ * JS) / WPB, WPB * 32, 0, stream>>>(
        hw, hu, Txw, Txb, X, mask, f);

    finalize_kernel<<<(n_out + 255) / 256, 256, 0, stream>>>(X, mask, f, out, n_out);
}
